// SelfAttentionLayer_13984413516355
// MI455X (gfx1250) — matrix operand order repaired and checked by reading, not compiled, not run on hardware
//
#include <hip/hip_runtime.h>

typedef unsigned short u16;
typedef unsigned int   u32;

typedef __attribute__((ext_vector_type(16))) __bf16 v16bf;
typedef __attribute__((ext_vector_type(8)))  __bf16 v8bf;
typedef __attribute__((ext_vector_type(8)))  float  v8f;

#define BB 4
#define SS 2048
#define DD 1024

__device__ __forceinline__ u16 f2bf(float f) {
  u32 u = __float_as_uint(f);
  u32 r = u + 0x7FFFu + ((u >> 16) & 1u);   // round-to-nearest-even
  return (u16)(r >> 16);
}
__device__ __forceinline__ float bf2f(u16 h) {
  return __uint_as_float(((u32)h) << 16);
}

// ---------------- CDNA5 async global->LDS path (guarded) ----------------
#if defined(__has_builtin)
#if __has_builtin(__builtin_amdgcn_global_load_async_to_lds_b128)
#define HAVE_ASYNC_LDS 1
#endif
#endif
#ifndef HAVE_ASYNC_LDS
#define HAVE_ASYNC_LDS 0
#endif

__device__ __forceinline__ void wait_async0() {
#if defined(__has_builtin) && __has_builtin(__builtin_amdgcn_s_wait_asynccnt)
  __builtin_amdgcn_s_wait_asynccnt(0);
#elif HAVE_ASYNC_LDS
  asm volatile("s_wait_asynccnt 0x0" ::: "memory");
#endif
}

#if HAVE_ASYNC_LDS
// Param 1 (global source) is a generic pointer to a 128-bit payload typed as
// int __attribute__((vector_size(16))) per the compiler's own diagnostic;
// param 2 (LDS dest) is the addrspace(3) flavor carrying the 32-bit LDS addr.
typedef int v4i __attribute__((vector_size(16)));
typedef __attribute__((address_space(3))) v4i v4i_lds;

__device__ __forceinline__ void gl_async_b128(const u16* g, u16* l) {
  // Generic flat pointers to __shared__ carry the LDS byte offset in the low
  // 32 bits (aperture in [63:32]); pun down to the AS3 pointer the builtin
  // expects for its destination operand.
  union { u16* p; v4i_lds* q; } b; b.p = l;
  __builtin_amdgcn_global_load_async_to_lds_b128((v4i*)g, b.q, 0, 0);
}
#endif

// Copy one 64-byte row (32 bf16) global -> LDS.
__device__ __forceinline__ void copy_row_64B(u16* lds, const u16* g) {
#if HAVE_ASYNC_LDS
#pragma unroll
  for (int u = 0; u < 4; ++u) gl_async_b128(g + u * 8, lds + u * 8);
#else
  const uint4* s = (const uint4*)g;
  uint4* d = (uint4*)lds;
  d[0] = s[0]; d[1] = s[1]; d[2] = s[2]; d[3] = s[3];
#endif
}

// ---- WMMA fragment loaders (LDS tiles are [128 rows][32 cols] bf16, pitch 32) ----
// A 16x32 bf16: lane<16 holds row m=lane, K=0..15;
// lane>=16 holds row m=lane-16, K=16..31 (matches frag_b).
__device__ __forceinline__ v16bf frag_a(const u16* lds, int mbase, int lane) {
  return *(const v16bf*)(lds + (mbase + (lane & 15)) * 32 + (lane >> 4) * 16);
}
// B 32x16 bf16: lanes 0-15 = col n, K=0..15; lanes 16-31 = col n, K=16..31.
// LDS tile stores B' as [n][k] so each lane reads 16 contiguous bf16 (32B).
__device__ __forceinline__ v16bf frag_b(const u16* lds, int nbase, int lane) {
  return *(const v16bf*)(lds + (nbase + (lane & 15)) * 32 + (lane >> 4) * 16);
}

// ============ Unified bf16 WMMA GEMM core, async double-buffered ============
// Block = 128 threads (4 waves, 2x2 grid of 64x64 wave tiles) computing a
// 128x128 tile of C = A(row-major [m][k]) * B'(stored [n][k])^T.
// lds: [2 buffers][A|B][128*32] bf16.
template <int NKT>
__device__ __forceinline__ void gemm_bf16_core(const u16* __restrict__ Ag, size_t pA, int m0,
                                               const u16* __restrict__ Bg, size_t pB, int n0,
                                               u16 (*lds)[2][4096], v8f (&acc)[4][4]) {
  const int t = threadIdx.x;
  const int lane = t & 31, wave = t >> 5;
  const int mb = (wave & 1) * 64, nbw = (wave >> 1) * 64;

  // prologue: issue tile 0 into buffer 0
  copy_row_64B(&lds[0][0][t * 32], Ag + (size_t)(m0 + t) * pA);
  copy_row_64B(&lds[0][1][t * 32], Bg + (size_t)(n0 + t) * pB);

#pragma unroll 1
  for (int kt = 0; kt < NKT; ++kt) {
    wait_async0();        // my async writes for tile kt landed in LDS
    __syncthreads();      // everyone's tile kt visible; buf[(kt+1)&1] free

    if (kt + 1 < NKT) {   // issue tile kt+1 while computing tile kt
      const int k1 = (kt + 1) * 32;
      u16 (&nxt)[2][4096] = lds[(kt + 1) & 1];
      copy_row_64B(&nxt[0][t * 32], Ag + (size_t)(m0 + t) * pA + k1);
      copy_row_64B(&nxt[1][t * 32], Bg + (size_t)(n0 + t) * pB + k1);
    }

    const u16* cA = lds[kt & 1][0];
    const u16* cB = lds[kt & 1][1];
    v16bf Af[4], Bf[4];
#pragma unroll
    for (int i = 0; i < 4; ++i) Af[i] = frag_a(cA, mb + 16 * i, lane);
#pragma unroll
    for (int j = 0; j < 4; ++j) Bf[j] = frag_b(cB, nbw + 16 * j, lane);
#pragma unroll
    for (int i = 0; i < 4; ++i)
#pragma unroll
      for (int j = 0; j < 4; ++j)
        acc[i][j] = __builtin_amdgcn_wmma_f32_16x16x32_bf16(
            false, Af[i], false, Bf[j], (short)0, acc[i][j], false, false);
  }
}

// ================= Kernel 0: fp32 -> bf16 bulk convert =================
__global__ void __launch_bounds__(256)
cvt_bf16(const float* __restrict__ in, u16* __restrict__ out, int n8) {
  const int i = blockIdx.x * 256 + threadIdx.x;
  if (i >= n8) return;
  const float4* s = (const float4*)in + (size_t)i * 2;
  const float4 a = s[0], b = s[1];
  u32* d = (u32*)out + (size_t)i * 4;
  d[0] = ((u32)f2bf(a.y) << 16) | f2bf(a.x);
  d[1] = ((u32)f2bf(a.w) << 16) | f2bf(a.z);
  d[2] = ((u32)f2bf(b.y) << 16) | f2bf(b.x);
  d[3] = ((u32)f2bf(b.w) << 16) | f2bf(b.z);
}

// ================= Kernel 1: QKV projection y = x @ W^T + b =================
// grid: (S/128, D/128, B*3), block: 128
__global__ void __launch_bounds__(128)
attn_qkv(const u16* __restrict__ xbf, const u16* __restrict__ Wbf,
         const float* __restrict__ bq, const float* __restrict__ bk,
         const float* __restrict__ bv,
         u16* __restrict__ Qbf, u16* __restrict__ Kbf, u16* __restrict__ Vt) {
  __shared__ alignas(64) u16 lds[2][2][4096];
  const int b = blockIdx.z / 3, which = blockIdx.z % 3;
  const int m0 = blockIdx.x * 128, n0 = blockIdx.y * 128;
  const float* bias = (which == 0) ? bq : (which == 1) ? bk : bv;

  v8f acc[4][4] = {};
  gemm_bf16_core<DD / 32>(xbf + (size_t)b * SS * DD, DD, m0,
                          Wbf + (size_t)which * DD * DD, DD, n0, lds, acc);

  const int lane = threadIdx.x & 31, wave = threadIdx.x >> 5;
  const int mb = (wave & 1) * 64, nbw = (wave >> 1) * 64;
#pragma unroll
  for (int j = 0; j < 4; ++j) {
    const int n = n0 + nbw + 16 * j + (lane & 15);
    const float bn = bias[n];
#pragma unroll
    for (int i = 0; i < 4; ++i) {
#pragma unroll
      for (int v = 0; v < 8; ++v) {
        const int m = m0 + mb + 16 * i + v + 8 * (lane >> 4);
        const u16 h = f2bf(acc[i][j][v] + bn);
        if (which == 2)      Vt[((size_t)b * DD + n) * SS + m] = h;  // V^T [b][d][s]
        else if (which == 1) Kbf[((size_t)b * SS + m) * DD + n] = h;
        else                 Qbf[((size_t)b * SS + m) * DD + n] = h;
      }
    }
  }
}

// ================= Kernel 2: scores = (Q @ K^T) / 32, bf16 out =================
// grid: (S/128, S/128, B), block: 128
__global__ void __launch_bounds__(128)
attn_scores(const u16* __restrict__ Qbf, const u16* __restrict__ Kbf,
            u16* __restrict__ SC) {
  __shared__ alignas(64) u16 lds[2][2][4096];
  const int b = blockIdx.z;
  const int m0 = blockIdx.x * 128, n0 = blockIdx.y * 128;

  v8f acc[4][4] = {};
  gemm_bf16_core<DD / 32>(Qbf + (size_t)b * SS * DD, DD, m0,
                          Kbf + (size_t)b * SS * DD, DD, n0, lds, acc);

  const int lane = threadIdx.x & 31, wave = threadIdx.x >> 5;
  const int mb = (wave & 1) * 64, nbw = (wave >> 1) * 64;
  const float scale = 1.0f / 32.0f;  // 1/sqrt(1024)
  u16* out = SC + (size_t)b * SS * SS;
#pragma unroll
  for (int i = 0; i < 4; ++i)
#pragma unroll
    for (int j = 0; j < 4; ++j) {
      const int n = n0 + nbw + 16 * j + (lane & 15);
#pragma unroll
      for (int v = 0; v < 8; ++v) {
        const int m = m0 + mb + 16 * i + v + 8 * (lane >> 4);
        out[(size_t)m * SS + n] = f2bf(acc[i][j][v] * scale);
      }
    }
}

// ================= Kernel 3: in-place row softmax over 2048 bf16 =================
// grid: (B*S), block: 256; each thread owns 8 contiguous elements (one uint4)
__global__ void __launch_bounds__(256)
attn_softmax(u16* __restrict__ SC) {
  __shared__ float red[256];
  const int t = threadIdx.x;
  u16* p = SC + (size_t)blockIdx.x * SS;

  uint4 raw = ((uint4*)p)[t];
  u32 w[4] = {raw.x, raw.y, raw.z, raw.w};
  float v[8];
#pragma unroll
  for (int u = 0; u < 4; ++u) {
    v[2 * u]     = bf2f((u16)(w[u] & 0xFFFFu));
    v[2 * u + 1] = bf2f((u16)(w[u] >> 16));
  }
  float mx = v[0];
#pragma unroll
  for (int i = 1; i < 8; ++i) mx = fmaxf(mx, v[i]);
  red[t] = mx; __syncthreads();
  for (int s = 128; s > 0; s >>= 1) {
    if (t < s) red[t] = fmaxf(red[t], red[t + s]);
    __syncthreads();
  }
  mx = red[0]; __syncthreads();

  float sum = 0.0f;
#pragma unroll
  for (int i = 0; i < 8; ++i) { v[i] = __expf(v[i] - mx); sum += v[i]; }
  red[t] = sum; __syncthreads();
  for (int s = 128; s > 0; s >>= 1) {
    if (t < s) red[t] += red[t + s];
    __syncthreads();
  }
  const float inv = 1.0f / red[0];

#pragma unroll
  for (int u = 0; u < 4; ++u)
    w[u] = ((u32)f2bf(v[2 * u + 1] * inv) << 16) | f2bf(v[2 * u] * inv);
  ((uint4*)p)[t] = make_uint4(w[0], w[1], w[2], w[3]);
}

// ================= Kernel 4: out = attn @ V (via V^T), fp32 out =================
// grid: (S/128, D/128, B), block: 128
__global__ void __launch_bounds__(128)
attn_out(const u16* __restrict__ SC, const u16* __restrict__ Vt,
         float* __restrict__ out) {
  __shared__ alignas(64) u16 lds[2][2][4096];
  const int b = blockIdx.z;
  const int m0 = blockIdx.x * 128, n0 = blockIdx.y * 128;

  v8f acc[4][4] = {};
  gemm_bf16_core<SS / 32>(SC + (size_t)b * SS * SS, SS, m0,
                          Vt + (size_t)b * DD * SS, SS, n0, lds, acc);

  const int lane = threadIdx.x & 31, wave = threadIdx.x >> 5;
  const int mb = (wave & 1) * 64, nbw = (wave >> 1) * 64;
#pragma unroll
  for (int i = 0; i < 4; ++i)
#pragma unroll
    for (int j = 0; j < 4; ++j) {
      const int n = n0 + nbw + 16 * j + (lane & 15);
#pragma unroll
      for (int v = 0; v < 8; ++v) {
        const int m = m0 + mb + 16 * i + v + 8 * (lane >> 4);
        out[((size_t)b * SS + m) * DD + n] = acc[i][j][v];
      }
    }
}

extern "C" void kernel_launch(void* const* d_in, const int* in_sizes, int n_in,
                              void* d_out, int out_size, void* d_ws, size_t ws_size,
                              hipStream_t stream) {
  const float* x  = (const float*)d_in[0];
  const float* Wq = (const float*)d_in[1];
  const float* bq = (const float*)d_in[2];
  const float* Wk = (const float*)d_in[3];
  const float* bk = (const float*)d_in[4];
  const float* Wv = (const float*)d_in[5];
  const float* bv = (const float*)d_in[6];
  float* out = (float*)d_out;

  // Workspace (bf16): Q | K | V^T | scores | x_bf16 | W_bf16[3]  ~= 103 MB
  u16* Qbf = (u16*)d_ws;
  u16* Kbf = Qbf + (size_t)BB * SS * DD;
  u16* Vt  = Kbf + (size_t)BB * SS * DD;
  u16* SC  = Vt  + (size_t)BB * SS * DD;           // B*S*S
  u16* xbf = SC  + (size_t)BB * SS * SS;
  u16* Wbf = xbf + (size_t)BB * SS * DD;           // [3][D][D]

  dim3 blk(128);
  {
    const int n8x = BB * SS * DD / 8;              // 1,048,576
    cvt_bf16<<<dim3((n8x + 255) / 256), dim3(256), 0, stream>>>(x, xbf, n8x);
    const int n8w = DD * DD / 8;                   // 131,072
    cvt_bf16<<<dim3((n8w + 255) / 256), dim3(256), 0, stream>>>(Wq, Wbf + 0 * (size_t)DD * DD, n8w);
    cvt_bf16<<<dim3((n8w + 255) / 256), dim3(256), 0, stream>>>(Wk, Wbf + 1 * (size_t)DD * DD, n8w);
    cvt_bf16<<<dim3((n8w + 255) / 256), dim3(256), 0, stream>>>(Wv, Wbf + 2 * (size_t)DD * DD, n8w);
  }
  attn_qkv<<<dim3(SS / 128, DD / 128, BB * 3), blk, 0, stream>>>(
      xbf, Wbf, bq, bk, bv, Qbf, Kbf, Vt);
  attn_scores<<<dim3(SS / 128, SS / 128, BB), blk, 0, stream>>>(Qbf, Kbf, SC);
  attn_softmax<<<dim3(BB * SS), dim3(256), 0, stream>>>(SC);
  attn_out<<<dim3(SS / 128, DD / 128, BB), blk, 0, stream>>>(SC, Vt, out);
}
